// ConvLRUModel_33243046871352
// MI455X (gfx1250) — matrix operand order repaired and checked by reading, not compiled, not run on hardware
//
#include <hip/hip_runtime.h>
#include <math.h>

// ---------------------------------------------------------------------------
// Problem constants: B=2, C=64, L=16, H=64, W=64, K=7, NG=4
//  - N_IMG = B*L = 32 images of [C=64][H*W=4096]
//  - conv folded with W_dc:  Weff[off][o][c] = sum_m W_dc[o,m] * W_sp[m,c,kh,kw]
//  - x pre-converted to bf16 in [n][h][w][c] layout (c fastest) so conv
//    staging is contiguous and WMMA B-fragments are contiguous LDS reads.
// ---------------------------------------------------------------------------

typedef unsigned short u16;
typedef __attribute__((ext_vector_type(16))) __bf16    v16bf;
typedef __attribute__((ext_vector_type(8)))  float     v8f;
typedef __attribute__((ext_vector_type(4)))  int       v4i;
typedef __attribute__((ext_vector_type(8)))  int       v8i;
typedef __attribute__((ext_vector_type(4)))  unsigned  v4u;

// Address-space-qualified types for gfx1250 async/TDM builtins
typedef __attribute__((address_space(1))) v4i  gas_v4i;   // global
typedef __attribute__((address_space(3))) v4i  las_v4i;   // LDS
typedef __attribute__((address_space(3))) char las_char;

union ABu { v16bf v; u16 s[16]; };

__device__ __forceinline__ u16 f2bf(float f) {
  union { float f; unsigned u; } x; x.f = f;
  unsigned u = x.u;
  u += 0x7fffu + ((u >> 16) & 1u);          // round-to-nearest-even
  return (u16)(u >> 16);
}

__device__ __forceinline__ v8f wmma_bf16(v16bf a, v16bf b, v8f c) {
  // v_wmma_f32_16x16x32_bf16 : D = A(16x32) * B(32x16) + C(16x16 f32)
  return __builtin_amdgcn_wmma_f32_16x16x32_bf16(
      /*neg_a=*/false, a, /*neg_b=*/false, b,
      /*c_mod=*/(short)0, c, /*reuse_a=*/false, /*reuse_b=*/false);
}

// Feature probes (clang 23.1 / amdgpu-toolchain confirmed)
#if defined(__has_builtin)
#  if __has_builtin(__builtin_amdgcn_tensor_load_to_lds)
#    define HAVE_TDM 1
#  endif
#  if __has_builtin(__builtin_amdgcn_global_load_async_to_lds_b128)
#    define HAVE_ASYNC_B128 1
#  endif
#  if __has_builtin(__builtin_amdgcn_s_wait_asynccnt)
#    define HAVE_WAIT_ASYNC 1
#  endif
#  if __has_builtin(__builtin_amdgcn_s_wait_tensorcnt)
#    define HAVE_WAIT_TENSOR 1
#  endif
#endif

// ---------------------------------------------------------------------------
// Prep 0: x (B,C,L,H,W) f32 -> x_bf [n=B*L][h][w][c] bf16 (transpose+convert)
//   One block per (n, h) row; LDS 64x64 tile keeps both sides coalesced.
// ---------------------------------------------------------------------------
__global__ __launch_bounds__(256) void prep_xbf_k(
    const float* __restrict__ x, u16* __restrict__ x_bf) {
  __shared__ u16 t[64][65];                        // +1 pad: no bank conflicts
  const int tid = threadIdx.x;
  const int n = blockIdx.x >> 6, h = blockIdx.x & 63;
  const int b = n >> 4, l = n & 15;
  for (int i = tid; i < 4096; i += 256) {          // read coalesced along w
    const int c = i >> 6, w = i & 63;
    t[w][c] = f2bf(x[(((size_t)b * 64 + c) * 16 + l) * 4096 + h * 64 + w]);
  }
  __syncthreads();
  u16* dst = x_bf + ((size_t)n * 4096 + h * 64) * 64;
  for (int i = tid; i < 4096; i += 256) {          // write coalesced along c
    const int w = i >> 6, c = i & 63;
    dst[w * 64 + c] = t[w][c];
  }
}

// ---------------------------------------------------------------------------
// Prep 1: fold W_dc into W_sp -> Weff bf16 [49][o=64][c=64]
// ---------------------------------------------------------------------------
__global__ __launch_bounds__(256) void prep_weff_k(
    const float* __restrict__ W_sp, const float* __restrict__ W_dc,
    u16* __restrict__ weff) {
  int i = blockIdx.x * 256 + threadIdx.x;
  if (i >= 49 * 64 * 64) return;
  int off = i >> 12;
  int o   = (i >> 6) & 63;
  int c   = i & 63;
  float s = 0.f;
#pragma unroll 8
  for (int m = 0; m < 64; ++m)
    s += W_dc[o * 64 + m] * W_sp[(m * 64 + c) * 49 + off];
  weff[off * 4096 + o * 64 + c] = f2bf(s);
}

// ---------------------------------------------------------------------------
// Prep 2: W_in (128x64) and W_out (64x64) -> bf16
// ---------------------------------------------------------------------------
__global__ __launch_bounds__(256) void prep_wio_k(
    const float* __restrict__ W_in, const float* __restrict__ W_out,
    u16* __restrict__ w_in_bf, u16* __restrict__ w_out_bf) {
  int i = blockIdx.x * 256 + threadIdx.x;
  if (i < 8192)       w_in_bf[i]         = f2bf(W_in[i]);
  else if (i < 12288) w_out_bf[i - 8192] = f2bf(W_out[i - 8192]);
}

// ---------------------------------------------------------------------------
// Conv kernel: one workgroup per (image n, output row h).
//   LDS slab: bf16 [r=7][w=64][c=64] = 56 KB; each kh-row is one contiguous
//   8 KB block in x_bf, staged by one TDM descriptor (source row edge-clamped).
//   8 waves: wave = (mTile 0..3, nPair 0..1), each computes 2 N-tiles of 16 px.
//   Implicit GEMM: 49 offsets x 2 k-chunks of 32 channels -> 196 WMMA / wave.
// ---------------------------------------------------------------------------
__global__ __launch_bounds__(256) void conv_k(
    const u16* __restrict__ x_bf, const u16* __restrict__ weff,
    const float* __restrict__ b_dc, float* __restrict__ y_ws) {
  extern __shared__ u16 smem_bf[];                 // 7*64*64 bf16 = 56 KB
  const int tid  = threadIdx.x;
  const int lane = tid & 31;
  const int wv   = tid >> 5;
  const int m16  = lane & 15;
  const int half = lane >> 4;
  const int n    = blockIdx.x >> 6;                // image 0..31
  const int h    = blockIdx.x & 63;                // output row

#if defined(HAVE_TDM)
  // ---- TDM staging: wave 0 issues 7 tensor tile loads (one per kh row) ----
  if (wv == 0) {
    const unsigned ldsBase = (unsigned)(size_t)(las_char*)smem_bf;
    const unsigned long long gImg =
        (unsigned long long)(x_bf + (size_t)n * 262144);   // 64h*64w*64c
    const v4i z4 = {0, 0, 0, 0};
    const v8i z8 = {0, 0, 0, 0, 0, 0, 0, 0};
    for (int r = 0; r < 7; ++r) {
      int hr = h + r - 3; hr = hr < 0 ? 0 : (hr > 63 ? 63 : hr);  // edge pad
      const unsigned long long ga = gImg + 2ull * (unsigned)(hr * 4096);
      // ---- D# group 0: count=1 | lds_addr | global_addr | type=2 ----
      v4u g0;
      g0.x = 1u;                                   // count=1, user mode
      g0.y = ldsBase + (unsigned)(r * 8192);       // LDS byte addr of slab r
      g0.z = (unsigned)(ga & 0xffffffffu);         // global_addr[31:0]
      g0.w = (unsigned)((ga >> 32) & 0x1ffffffu)   // global_addr[56:32]
             | 0x80000000u;                        // type=2 ("image")
      // ---- D# group 1: 1-line tile of 4096 bf16 (contiguous [w][c] plane) --
      v8i g1;
      g1[0] = 0x00010000;            // wg_mask=0 (no cluster), data_size=1 (2B)
      g1[1] = (int)(4096u << 16);    // tensor_dim0[15:0]=4096
      g1[2] = (int)(1u << 16);       // tensor_dim0 hi=0 | tensor_dim1[15:0]=1
      g1[3] = (int)(4096u << 16);    // tensor_dim1 hi=0 | tile_dim0=4096
      g1[4] = 1;                     // tile_dim1=1 | tile_dim2=0
      g1[5] = 4096;                  // tensor_dim0_stride[31:0]
      g1[6] = 0;
      g1[7] = 0;
      __builtin_amdgcn_tensor_load_to_lds(g0, g1, z4, z4, z8, 0);
    }
#  if defined(HAVE_WAIT_TENSOR)
    __builtin_amdgcn_s_wait_tensorcnt(0);
#  else
    asm volatile("s_wait_tensorcnt 0" ::: "memory");
#  endif
  }
#else
  // ---- fallback staging: per-thread 16B chunks ----
  for (int i = tid; i < 7 * 512; i += 256) {       // 7 rows * 512 chunks
    const int r = i >> 9;
    const int q = i & 511;                         // 16B = 8 bf16 elems
    int hr = h + r - 3; hr = hr < 0 ? 0 : (hr > 63 ? 63 : hr);   // edge pad
    const u16* gp = x_bf + ((size_t)n * 262144 + (size_t)hr * 4096 + q * 8);
    u16* lp = smem_bf + (r * 4096 + q * 8);
#  if defined(HAVE_ASYNC_B128)
    __builtin_amdgcn_global_load_async_to_lds_b128((gas_v4i*)gp, (las_v4i*)lp, 0, 0);
#  else
    *(v4i*)lp = *(const v4i*)gp;
#  endif
  }
#  if defined(HAVE_ASYNC_B128)
#    if defined(HAVE_WAIT_ASYNC)
  __builtin_amdgcn_s_wait_asynccnt(0);
#    else
  asm volatile("s_wait_asynccnt 0" ::: "memory");
#    endif
#  endif
#endif
  __syncthreads();

  const int mT  = wv & 3;                          // out-channel tile 0..3
  const int nt0 = (wv >> 2) * 2;                   // first pixel tile (0 or 2)

  v8f acc0 = {};
  v8f acc1 = {};

#pragma unroll 1
  for (int off = 0; off < 49; ++off) {
    const int dh = off / 7;                        // LDS row index (kh)
    const int dw = off - dh * 7;                   // kw
#pragma unroll
    for (int kq = 0; kq < 2; ++kq) {
      // A fragment: Weff[off][o = mT*16 + m][c = kq*32 ...], contiguous k order c = kq*32 + 16*half + s, same as B
      ABu a;
      const u16* wp = weff + off * 4096 + (mT * 16 + m16) * 64 + kq * 32 + half * 16;
#pragma unroll
      for (int j = 0; j < 8; ++j) { a.s[j] = wp[j]; a.s[8 + j] = wp[8 + j]; }

      // B fragments: 16 contiguous bf16 channels at the wrapped pixel column
#pragma unroll
      for (int t = 0; t < 2; ++t) {
        const int wcol = (nt0 + t) * 16 + m16;
        const int srcw = (wcol + dw + 61) & 63;    // wrap pad: w + kw - 3 mod 64
        const u16* bp  = smem_bf + ((dh * 64 + srcw) * 64 + kq * 32 + half * 16);
        ABu bf;
#pragma unroll
        for (int e = 0; e < 16; ++e) bf.s[e] = bp[e];
        if (t == 0) acc0 = wmma_bf16(a.v, bf.v, acc0);
        else        acc1 = wmma_bf16(a.v, bf.v, acc1);
      }
    }
  }

  // ---- store D fragments + b_dc :  y_ws[n][o][h*64 + pix] ----
#pragma unroll
  for (int t = 0; t < 2; ++t) {
    const int pix = h * 64 + (nt0 + t) * 16 + m16;
#pragma unroll
    for (int r = 0; r < 8; ++r) {
      const int o = mT * 16 + half * 8 + r;
      const float v = (t == 0 ? acc0[r] : acc1[r]) + b_dc[o];
      y_ws[((size_t)n * 64 + o) * 4096 + pix] = v;
    }
  }
}

// ---------------------------------------------------------------------------
// GroupNorm stats: one block per (n, g) -> mean & rsqrt(var+eps) over 16c*4096px
// ---------------------------------------------------------------------------
__global__ __launch_bounds__(256) void gn_k(
    const float* __restrict__ y_ws, float* __restrict__ mu, float* __restrict__ rs) {
  __shared__ float r1[256], r2[256];
  const int tid = threadIdx.x;
  const int n = blockIdx.x >> 2, g = blockIdx.x & 3;
  const float* base = y_ws + (size_t)n * 64 * 4096 + (size_t)g * 16 * 4096;
  float s1 = 0.f, s2 = 0.f;
  for (int i = tid; i < 65536; i += 256) { float v = base[i]; s1 += v; s2 += v * v; }
  r1[tid] = s1; r2[tid] = s2; __syncthreads();
  for (int s = 128; s > 0; s >>= 1) {
    if (tid < s) { r1[tid] += r1[tid + s]; r2[tid] += r2[tid + s]; }
    __syncthreads();
  }
  if (tid == 0) {
    float mean = r1[0] * (1.f / 65536.f);
    float var  = r2[0] * (1.f / 65536.f) - mean * mean;
    mu[blockIdx.x] = mean;
    rs[blockIdx.x] = rsqrtf(var + 1e-5f);
  }
}

// ---------------------------------------------------------------------------
// Fused normalize + W_in GEMM + gate + W_out GEMM + residual. WMMA per wave,
// each wave owns 16 pixels. Writes u in-place into y_ws.
// ---------------------------------------------------------------------------
__global__ __launch_bounds__(256) void mlp_k(
    const float* __restrict__ x, float* __restrict__ y_ws,
    const float* __restrict__ mu, const float* __restrict__ rs,
    const float* __restrict__ gn_g, const float* __restrict__ gn_b,
    const u16* __restrict__ w_in_bf, const float* __restrict__ b_in,
    const u16* __restrict__ w_out_bf, const float* __restrict__ b_out) {
  __shared__ u16 sb[8][2048];                      // per wave: yn[16][64] + z[16][64]
  const int tid  = threadIdx.x;
  const int lane = tid & 31;
  const int wv   = tid >> 5;
  const int m16  = lane & 15;
  const int half = lane >> 4;
  u16* ynb = sb[wv];
  u16* zb  = sb[wv] + 1024;

  const int task    = blockIdx.x * 8 + wv;         // 8192 tasks
  const int n       = task >> 8;                   // image
  const int pixBase = (task & 255) * 16;
  const int bImg = n >> 4, lImg = n & 15;

  // ---- stage normalized activations: yn[pix][c] bf16 ----
  for (int i = lane; i < 1024; i += 32) {
    const int p = i >> 6, c = i & 63, g = c >> 4;
    const float v  = y_ws[((size_t)n * 64 + c) * 4096 + pixBase + p];
    const float yn = (v - mu[n * 4 + g]) * rs[n * 4 + g] * gn_g[c] + gn_b[c];
    ynb[p * 64 + c] = f2bf(yn);
  }
  __syncthreads();

  // ---- h = W_in @ yn : M=128 (8 tiles), K=64 (2 chunks), N=16 px ----
  v8f hacc[8];
#pragma unroll
  for (int mt = 0; mt < 8; ++mt) {
    v8f acc = {};
#pragma unroll
    for (int kq = 0; kq < 2; ++kq) {
      ABu a;
      const u16* wp = w_in_bf + (mt * 16 + m16) * 64 + kq * 32 + half * 16;
#pragma unroll
      for (int j = 0; j < 8; ++j) { a.s[j] = wp[j]; a.s[8 + j] = wp[8 + j]; }
      ABu bf;
      const u16* yp = ynb + m16 * 64 + kq * 32 + half * 16;
#pragma unroll
      for (int e = 0; e < 16; ++e) bf.s[e] = yp[e];
      acc = wmma_bf16(a.v, bf.v, acc);
    }
    hacc[mt] = acc;
  }

  // ---- gate z = x1 * sigmoid(x2), store z[pix][c] for the next GEMM ----
#pragma unroll
  for (int f = 0; f < 4; ++f) {
#pragma unroll
    for (int r = 0; r < 8; ++r) {
      const int row = f * 16 + half * 8 + r;       // 0..63
      const float hv = hacc[f][r]     + b_in[row];
      const float gv = hacc[f + 4][r] + b_in[64 + row];
      const float z  = hv * (1.f / (1.f + expf(-gv)));
      zb[m16 * 64 + row] = f2bf(z);
    }
  }
  __syncthreads();

  // ---- u = W_out @ z + b_out + xf  (residual), in-place into y_ws ----
#pragma unroll
  for (int mt = 0; mt < 4; ++mt) {
    v8f acc = {};
#pragma unroll
    for (int kq = 0; kq < 2; ++kq) {
      ABu a;
      const u16* wp = w_out_bf + (mt * 16 + m16) * 64 + kq * 32 + half * 16;
#pragma unroll
      for (int j = 0; j < 8; ++j) { a.s[j] = wp[j]; a.s[8 + j] = wp[8 + j]; }
      ABu bf;
      const u16* zp = zb + m16 * 64 + kq * 32 + half * 16;
#pragma unroll
      for (int e = 0; e < 16; ++e) bf.s[e] = zp[e];
      acc = wmma_bf16(a.v, bf.v, acc);
    }
#pragma unroll
    for (int r = 0; r < 8; ++r) {
      const int o   = mt * 16 + half * 8 + r;
      const int pix = pixBase + m16;
      const float xf = x[(((size_t)bImg * 64 + o) * 16 + lImg) * 4096 + pix];
      y_ws[((size_t)n * 64 + o) * 4096 + pix] = acc[r] + b_out[o] + xf;
    }
  }
}

// ---------------------------------------------------------------------------
// LRU scan over L=16: one thread per (b, c, pixel). Memory-bound elementwise.
// ---------------------------------------------------------------------------
__global__ __launch_bounds__(256) void scan_k(
    const float* __restrict__ u_ws, const float* __restrict__ dt,
    const float* __restrict__ nu_log, const float* __restrict__ theta_log,
    const float* __restrict__ c_re, const float* __restrict__ c_im,
    const float* __restrict__ d_skip, float* __restrict__ out) {
  const int t   = blockIdx.x * 256 + threadIdx.x;  // 524288 threads
  const int pix = t & 4095;
  const int c   = (t >> 12) & 63;
  const int b   = t >> 18;
  const float nu  = expf(nu_log[c]);
  const float th  = expf(theta_log[c]);
  const float crv = c_re[c], civ = c_im[c], dsv = d_skip[c];
  float hre = 0.f, him = 0.f;
#pragma unroll 1
  for (int l = 0; l < 16; ++l) {
    const float dtv   = dt[b * 16 + l];
    const float decay = expf(-nu * dtv);
    const float ph    = th * dtv;
    const float lr    = decay * cosf(ph);
    const float li    = decay * sinf(ph);
    const float gm    = sqrtf(fmaxf(1.f - decay * decay, 1e-6f));
    const float u     = u_ws[(((size_t)(b * 16 + l)) * 64 + c) * 4096 + pix];
    const float nr = lr * hre - li * him + gm * u;
    const float ni = li * hre + lr * him;
    hre = nr; him = ni;
    out[(((size_t)(b * 64 + c)) * 16 + l) * 4096 + pix] = crv * nr + civ * ni + dsv * u;
  }
}

// ---------------------------------------------------------------------------
// Launch
// ---------------------------------------------------------------------------
extern "C" void kernel_launch(void* const* d_in, const int* in_sizes, int n_in,
                              void* d_out, int out_size, void* d_ws, size_t ws_size,
                              hipStream_t stream) {
  (void)in_sizes; (void)n_in; (void)out_size; (void)ws_size;
  const float* x         = (const float*)d_in[0];
  const float* dt        = (const float*)d_in[1];
  const float* W_sp      = (const float*)d_in[2];
  const float* W_dc      = (const float*)d_in[3];
  const float* b_dc      = (const float*)d_in[4];
  const float* gn_g      = (const float*)d_in[5];
  const float* gn_b      = (const float*)d_in[6];
  const float* W_in      = (const float*)d_in[7];
  const float* b_in      = (const float*)d_in[8];
  const float* W_out     = (const float*)d_in[9];
  const float* b_out     = (const float*)d_in[10];
  const float* nu_log    = (const float*)d_in[11];
  const float* theta_log = (const float*)d_in[12];
  const float* c_re      = (const float*)d_in[13];
  const float* c_im      = (const float*)d_in[14];
  const float* d_skip    = (const float*)d_in[15];
  float* out = (float*)d_out;

  char* ws = (char*)d_ws;
  float* y_ws    = (float*)(ws);                   // 33,554,432 B (y, then u in-place)
  u16*  x_bf     = (u16*)(ws + 33554432);          // 16,777,216 B  [n][h][w][c] bf16
  u16*  weff     = (u16*)(ws + 50331648);          //    401,408 B  [49][o][c] bf16
  u16*  w_in_bf  = (u16*)(ws + 50733056);          //     16,384 B
  u16*  w_out_bf = (u16*)(ws + 50749440);          //      8,192 B
  float* mu      = (float*)(ws + 50757632);        //        512 B
  float* rs      = (float*)(ws + 50758144);        //        512 B

  prep_xbf_k<<<2048, 256, 0, stream>>>(x, x_bf);
  prep_weff_k<<<784, 256, 0, stream>>>(W_sp, W_dc, weff);
  prep_wio_k<<<48, 256, 0, stream>>>(W_in, W_out, w_in_bf, w_out_bf);
  conv_k<<<2048, 256, 7 * 64 * 64 * sizeof(u16), stream>>>(x_bf, weff, b_dc, y_ws);
  gn_k<<<128, 256, 0, stream>>>(y_ws, mu, rs);
  mlp_k<<<1024, 256, 0, stream>>>(x, y_ws, mu, rs, gn_g, gn_b, w_in_bf, b_in, w_out_bf, b_out);
  scan_k<<<2048, 256, 0, stream>>>(y_ws, dt, nu_log, theta_log, c_re, c_im, d_skip, out);
}
